// DimeNetWrapper_83820581749197
// MI455X (gfx1250) — matrix operand order repaired and checked by reading, not compiled, not run on hardware
//
#include <hip/hip_runtime.h>

// ---------------- problem constants (fixed by wrapper) ----------------
#define HD     128
#define OUTD   14
#define NS_    7
#define NR_    6
#define NBIL_  8
#define NG_    32
#define NNODE_ 512
#define NE_    7680
#define NT_    107520
#define TPE_   14
#define EPG_   240

typedef _Float16 half_t;
typedef __attribute__((ext_vector_type(16))) _Float16 v16h;
typedef __attribute__((ext_vector_type(8)))  _Float16 v8h;
typedef __attribute__((ext_vector_type(8)))  float    v8f;

#define EPI_STORE          0
#define EPI_BIAS_SILU      1
#define EPI_BIAS_SILU_MUL  2
#define EPI_ADD_AUX        3
#define EPI_BIAS_SILU_ADD  4
#define EPI_ACCUM          5

__device__ __forceinline__ float silu_f(float v){ return v / (1.0f + expf(-v)); }

// ---------------- init: spherical-Bessel zeros / norms / legendre coefs ----
__device__ double sph_jn_d(int l, double x){
  double j0 = sin(x)/x;
  if (l == 0) return j0;
  double jm = j0;
  double jc = sin(x)/(x*x) - cos(x)/x;
  for (int i = 1; i < l; i++){ double t = (2.0*i+1.0)/x*jc - jm; jm = jc; jc = t; }
  return jc;
}

__global__ void k_init(float* cst){
  if (threadIdx.x != 0 || blockIdx.x != 0) return;
  const double PI = 3.14159265358979323846;
  double z[NS_][NS_+NR_];
  const int m = NS_ + NR_;
  for (int i=0;i<m;i++) z[0][i] = (i+1)*PI;
  for (int l=1;l<NS_;l++){
    for (int i=0;i<m-l;i++){
      double a=z[l-1][i], b=z[l-1][i+1];
      for (int it=0; it<90; it++){
        double c = 0.5*(a+b);
        if (sph_jn_d(l,a)*sph_jn_d(l,c) <= 0.0) b=c; else a=c;
      }
      z[l][i] = 0.5*(a+b);
    }
  }
  for (int l=0;l<NS_;l++)
    for (int r=0;r<NR_;r++){
      double zz = z[l][r];
      cst[l*NR_+r]      = (float)zz;                                   // Z_J
      cst[42 + l*NR_+r] = (float)(sqrt(2.0)/fabs(sph_jn_d(l+1, zz)));  // NORM_J
    }
  for (int l=0;l<NS_;l++) cst[84+l] = (float)sqrt((2.0*l+1.0)/(4.0*PI)); // legendre coef
}

// ---------------- fp32 spherical bessel (matches reference, series for x<0.5)
__device__ float sph_jn_f(int l, float x){
  bool sm = x < 0.5f;
  float xr = sm ? 1.0f : x;
  float jc = sinf(xr)/xr;
  if (l > 0){
    float jm = jc;
    jc = sinf(xr)/(xr*xr) - cosf(xr)/xr;
    for (int i=1;i<l;i++){ float t = (2.0f*i+1.0f)/xr*jc - jm; jm=jc; jc=t; }
  }
  float xs = sm ? x : 0.0f;
  float df = 1.0f;
  for (int i=1;i<=l;i++) df *= (2.0f*i+1.0f);
  float xl = 1.0f;
  for (int i=0;i<l;i++) xl *= xs;
  float x2 = xs*xs;
  float ser = xl/df * (1.0f - x2/(2.0f*(2*l+3)) + x2*x2/(8.0f*(2*l+3)*(2*l+5)));
  return sm ? ser : jc;
}

// ---------------- per-edge geometry: rbf[E,6], sb[E,42] ----------------
__global__ void k_edge(const float* __restrict__ pos, const int* __restrict__ src,
                       const int* __restrict__ dst, const float* __restrict__ freq,
                       const float* __restrict__ cst,
                       float* __restrict__ rbf, float* __restrict__ sb){
  int e = blockIdx.x*blockDim.x + threadIdx.x;
  if (e >= NE_) return;
  int s = src[e], d = dst[e];
  float vx = pos[d*3+0]-pos[s*3+0];
  float vy = pos[d*3+1]-pos[s*3+1];
  float vz = pos[d*3+2]-pos[s*3+2];
  float dist = sqrtf(vx*vx+vy*vy+vz*vz);
  float x = dist * 0.1f;                              // dist / CUTOFF
  float x2 = x*x; float x5 = x2*x2*x;
  float env = (x < 1.0f) ? (1.0f/x - 28.0f*x5 + 48.0f*x5*x - 21.0f*x5*x2) : 0.0f;
  for (int r=0;r<NR_;r++) rbf[e*NR_+r] = env * sinf(freq[r]*x);
  for (int l=0;l<NS_;l++)
    for (int r=0;r<NR_;r++)
      sb[e*42 + l*NR_ + r] = cst[42 + l*NR_ + r] * sph_jn_f(l, cst[l*NR_+r]*x) * env;
}

// ---------------- per-triplet cos(angle) ----------------
__global__ void k_cth(const float* __restrict__ pos, const int* __restrict__ src,
                      const int* __restrict__ dst, const int* __restrict__ ikj,
                      const int* __restrict__ iji, float* __restrict__ cth){
  int t = blockIdx.x*blockDim.x + threadIdx.x;
  if (t >= NT_) return;
  int eji = iji[t], ekj = ikj[t];
  int j = src[eji], i = dst[eji], k = src[ekj];
  float ax = pos[i*3+0]-pos[j*3+0], ay = pos[i*3+1]-pos[j*3+1], az = pos[i*3+2]-pos[j*3+2];
  float bx = pos[k*3+0]-pos[j*3+0], by = pos[k*3+1]-pos[j*3+1], bz = pos[k*3+2]-pos[j*3+2];
  float dp = ax*bx+ay*by+az*bz;
  float cx = ay*bz-az*by, cy = az*bx-ax*bz, cz = ax*by-ay*bx;
  float cr2 = cx*cx+cy*cy+cz*cz;
  cth[t] = dp / sqrtf(dp*dp + cr2);                   // cos(atan2(|cross|, dot))
}

// ---------------- per-interaction: sbf_h[T,8] = (sb[idx_kj]*legendre) @ lin_sbf
__global__ void k_sbfh(const float* __restrict__ cth, const int* __restrict__ ikj,
                       const float* __restrict__ sb, const float* __restrict__ lin_sbf,
                       const float* __restrict__ cst, float* __restrict__ sbfh){
  int t = blockIdx.x*blockDim.x + threadIdx.x;
  if (t >= NT_) return;
  float c = cth[t];
  float leg[NS_];
  float p0 = 1.0f, p1 = c;
  leg[0] = cst[84]; leg[1] = cst[85]*c;
  for (int l=2;l<NS_;l++){ float p = ((2.0f*l-1.0f)*c*p1 - (l-1.0f)*p0)/(float)l; p0=p1; p1=p; leg[l]=cst[84+l]*p; }
  const float* sbr = sb + (size_t)ikj[t]*42;
  float acc[NBIL_];
  for (int j=0;j<NBIL_;j++) acc[j]=0.0f;
  for (int l=0;l<NS_;l++){
    float lg = leg[l];
    for (int r=0;r<NR_;r++){
      float v = sbr[l*NR_+r]*lg;
      const float* w = lin_sbf + (l*NR_+r)*NBIL_;
      for (int j=0;j<NBIL_;j++) acc[j] += v*w[j];
    }
  }
  for (int j=0;j<NBIL_;j++) sbfh[(size_t)t*NBIL_+j] = acc[j];
}

// ---------------- rbf_h[e,h] = rbf[e,:6] @ W[6,128] ----------------
__global__ void k_rbfh(const float* __restrict__ rbf, const float* __restrict__ w,
                       float* __restrict__ out){
  int e = blockIdx.x, hh = threadIdx.x;
  float a = 0.0f;
  for (int r=0;r<NR_;r++) a += rbf[e*NR_+r]*w[r*HD+hh];
  out[(size_t)e*HD+hh] = a;
}

// ---------------- out-block t: tout = (rbf@W) * x ----------------
__global__ void k_outt(const float* __restrict__ rbf, const float* __restrict__ w,
                       const float* __restrict__ x, float* __restrict__ out){
  int e = blockIdx.x, hh = threadIdx.x;
  float a = 0.0f;
  for (int r=0;r<NR_;r++) a += rbf[e*NR_+r]*w[r*HD+hh];
  size_t idx = (size_t)e*HD+hh;
  out[idx] = a * x[idx];
}

// ---------------- embedding concat (f16): [hz[dst], hz[src], rbf_e] ------
__global__ void k_embcat(const float* __restrict__ rbf, const float* __restrict__ wr,
                         const float* __restrict__ br, const float* __restrict__ table,
                         const int* __restrict__ z, const int* __restrict__ src,
                         const int* __restrict__ dst, half_t* __restrict__ cat){
  int e = blockIdx.x, hh = threadIdx.x;
  float a = br[hh];
  for (int r=0;r<NR_;r++) a += rbf[e*NR_+r]*wr[r*HD+hh];
  a = silu_f(a);
  size_t base = (size_t)e*(3*HD);
  cat[base + 2*HD + hh] = (half_t)a;
  cat[base + hh]        = (half_t)table[(size_t)z[dst[e]]*HD + hh];
  cat[base + HD + hh]   = (half_t)table[(size_t)z[src[e]]*HD + hh];
}

// -------- per-edge bilinear A: A[e, j*128+l] = sum_t sbf_h[t,j]*xk[idx_kj[t],l]
__global__ void k_abil(const float* __restrict__ xk, const float* __restrict__ sbfh,
                       const int* __restrict__ ikj, const int* __restrict__ iji,
                       half_t* __restrict__ abil){
  __shared__ float lx[TPE_*HD];
  __shared__ float lsh[TPE_*NBIL_];
  int e = blockIdx.x, tid = threadIdx.x;
  int base = e*TPE_;
  for (int tt=0; tt<TPE_; tt++){
    int ek = ikj[base+tt];
    lx[tt*HD + tid] = xk[(size_t)ek*HD + tid];
  }
  if (tid < TPE_*NBIL_) lsh[tid] = sbfh[(size_t)base*NBIL_ + tid];
  __syncthreads();
  int erow = iji[base];                       // == e by construction
  for (int j=0;j<NBIL_;j++){
    float a = 0.0f;
    for (int tt=0; tt<TPE_; tt++) a += lsh[tt*NBIL_+j]*lx[tt*HD+tid];
    abil[(size_t)erow*1024 + j*HD + tid] = (half_t)a;
  }
}

// ---------------- dst-gather segment sum over complete graph -------------
__global__ void k_gather(const float* __restrict__ tout, float* __restrict__ y){
  int v = blockIdx.x, hh = threadIdx.x;
  int g = v >> 4, i = v & 15;
  float acc = 0.0f;
  for (int j=0;j<16;j++){
    if (j == i) continue;
    int e = g*EPG_ + j*15 + i - (i > j ? 1 : 0);
    acc += tout[(size_t)e*HD + hh];
  }
  y[(size_t)v*HD + hh] = acc;
}

// ---- weight conversion f32 -> f16, K-chunked layout Bt[k/32][n][k%32] ----
__global__ void k_wconv(const float* __restrict__ src, half_t* __restrict__ dst,
                        int K, int N, int Np){
  int idx = blockIdx.x*blockDim.x + threadIdx.x;
  if (idx >= K*Np) return;
  int k = idx / Np, n = idx % Np;
  half_t v = (n < N) ? (half_t)src[(size_t)k*N + n] : (half_t)0.0f;
  dst[((size_t)(k >> 5)*Np + n)*32 + (k & 31)] = v;
}

// Wr chunked: k index kk = j*128+l (K=1024), col i (N=128)
__global__ void k_wconv_bil(const float* __restrict__ src, half_t* __restrict__ dst){
  int idx = blockIdx.x*blockDim.x + threadIdx.x;   // 131072 threads
  int kk = idx >> 7, i = idx & 127;
  int j = kk >> 7, l = kk & 127;
  dst[((size_t)(kk >> 5)*HD + i)*32 + (kk & 31)] = (half_t)src[(size_t)i*1024 + j*HD + l];
}

__global__ void k_zero(float* __restrict__ p, int n){
  int idx = blockIdx.x*blockDim.x + threadIdx.x;
  if (idx < n) p[idx] = 0.0f;
}

// ---------------- final per-graph reduction ----------------
__global__ void k_graphsum(const float* __restrict__ P, float* __restrict__ out){
  int idx = blockIdx.x*blockDim.x + threadIdx.x;
  if (idx >= NG_*OUTD) return;
  int g = idx / OUTD, o = idx % OUTD;
  float acc = 0.0f;
  for (int i=0;i<16;i++) acc += P[(size_t)(g*16+i)*16 + o];
  out[idx] = acc;
}

// ================= generic WMMA GEMM (templated) =================
// C[M,N] = epi( A[M,K] @ B[K,N] ); A f32 or f16; B f16 in K-chunked layout
// Bt[(k/32)*N + n][k%32]. Block = 128 threads (4 waves), tile 64 x (NSUB*16).
template<int NSUB, int AHALF, int MODE>
__global__ __launch_bounds__(128)
void k_gemm(const void* __restrict__ A, const half_t* __restrict__ B,
            float* __restrict__ C, int M, int N, int K,
            const float* __restrict__ bias, const float* __restrict__ aux){
  __shared__ v16h lAv[128];           // 64 rows x 32 halfs
  __shared__ v16h lBv[NSUB*32];       // NT cols x 32 halfs (k-contig per col)
  half_t* lA = (half_t*)lAv;
  half_t* lB = (half_t*)lBv;
  const int NT = NSUB*16;
  int tid  = threadIdx.x;
  int wave = tid >> 5, lane = tid & 31;
  int bm = blockIdx.x * 64;
  int bn = blockIdx.y * NT;
  const float*  Af = (const float*)A;
  const half_t* Ah = (const half_t*)A;

  v8f acc[NSUB];
  #pragma unroll
  for (int s=0;s<NSUB;s++)
    #pragma unroll
    for (int r=0;r<8;r++) acc[s][r] = 0.0f;

  const int ar = tid >> 1, ac0 = (tid & 1) << 4;      // A staging: 16 contig elems
  const int frow = wave*16 + (lane & 15);
  const int fkb  = (lane >> 4) << 3;
  const int fks  = (lane >> 4) << 4;

  for (int kc = 0; kc < K; kc += 32){
    // ---- stage A (64x32), vectorized, f32->f16 convert if needed ----
    {
      size_t gbase = (size_t)(bm + ar) * K + kc + ac0;
      v16h v;
      if (AHALF){
        const v8h* g = (const v8h*)(Ah + gbase);
        v8h x0 = g[0], x1 = g[1];
        v = __builtin_shufflevector(x0, x1, 0,1,2,3,4,5,6,7,8,9,10,11,12,13,14,15);
      } else {
        const float4* g = (const float4*)(Af + gbase);
        #pragma unroll
        for (int q=0;q<4;q++){
          float4 f = g[q];
          v[q*4+0]=(half_t)f.x; v[q*4+1]=(half_t)f.y;
          v[q*4+2]=(half_t)f.z; v[q*4+3]=(half_t)f.w;
        }
      }
      lAv[tid] = v;   // row ar, cols ac0..ac0+15
    }
    // ---- stage B: contiguous copy of this K-chunk's [bn, bn+NT) columns ----
    {
      const v16h* gb = (const v16h*)(B + ((size_t)(kc >> 5)*N + bn)*32);
      #pragma unroll
      for (int i = tid; i < NT*2; i += 128) lBv[i] = gb[i];
    }
    __syncthreads();

    // A fragment: lanes<16 hold K {0..15}; lanes>=16 hold {16..31}
    v16h af = *(const v16h*)&lA[frow*32 + fks];

    #pragma unroll
    for (int s=0; s<NSUB; s++){
      // B fragment: lane n = lane&15; K = 16*(lane>=16) + e
      int n = (s << 4) + (lane & 15);
      v16h bf = *(const v16h*)&lB[n*32 + fks];
      acc[s] = __builtin_amdgcn_wmma_f32_16x16x32_f16(
          false, af, false, bf, (short)0, acc[s], false, false);
    }
    __syncthreads();
  }

  // epilogue: C layout lane n=lane&15, m = 8*(lane>=16)+r
  int n_l  = lane & 15;
  int mhi  = (lane >> 4) << 3;
  int row0 = bm + wave*16;
  #pragma unroll
  for (int s=0; s<NSUB; s++){
    int col = bn + (s << 4) + n_l;
    float bv = (MODE==EPI_BIAS_SILU || MODE==EPI_BIAS_SILU_MUL || MODE==EPI_BIAS_SILU_ADD)
             ? bias[col] : 0.0f;
    #pragma unroll
    for (int r=0; r<8; r++){
      int row = row0 + mhi + r;
      size_t idx = (size_t)row * N + col;
      float v = acc[s][r];
      if      (MODE==EPI_STORE)         C[idx] = v;
      else if (MODE==EPI_BIAS_SILU)     C[idx] = silu_f(v + bv);
      else if (MODE==EPI_BIAS_SILU_MUL) C[idx] = silu_f(v + bv) * aux[idx];
      else if (MODE==EPI_ADD_AUX)       C[idx] = v + aux[idx];
      else if (MODE==EPI_BIAS_SILU_ADD) C[idx] = aux[idx] + silu_f(v + bv);
      else if (MODE==EPI_ACCUM)         C[idx] += v;
    }
  }
}

// ================= host side =================
static inline void launch_gemm(hipStream_t st, const void* A, int a_half,
                               const half_t* B, float* C, int M, int N, int K,
                               const float* bias, const float* aux, int mode){
  if (N >= 64){
    dim3 grid((unsigned)(M/64), (unsigned)(N/64));
    if (a_half){
      if (mode == EPI_BIAS_SILU)
        k_gemm<4,1,EPI_BIAS_SILU><<<grid,128,0,st>>>(A,B,C,M,N,K,bias,aux);
      else if (mode == EPI_ADD_AUX)
        k_gemm<4,1,EPI_ADD_AUX><<<grid,128,0,st>>>(A,B,C,M,N,K,bias,aux);
      else
        k_gemm<4,1,EPI_STORE><<<grid,128,0,st>>>(A,B,C,M,N,K,bias,aux);
    } else {
      switch (mode){
        case EPI_BIAS_SILU:
          k_gemm<4,0,EPI_BIAS_SILU><<<grid,128,0,st>>>(A,B,C,M,N,K,bias,aux); break;
        case EPI_BIAS_SILU_MUL:
          k_gemm<4,0,EPI_BIAS_SILU_MUL><<<grid,128,0,st>>>(A,B,C,M,N,K,bias,aux); break;
        case EPI_ADD_AUX:
          k_gemm<4,0,EPI_ADD_AUX><<<grid,128,0,st>>>(A,B,C,M,N,K,bias,aux); break;
        case EPI_BIAS_SILU_ADD:
          k_gemm<4,0,EPI_BIAS_SILU_ADD><<<grid,128,0,st>>>(A,B,C,M,N,K,bias,aux); break;
        default:
          k_gemm<4,0,EPI_STORE><<<grid,128,0,st>>>(A,B,C,M,N,K,bias,aux); break;
      }
    }
  } else {
    dim3 grid((unsigned)(M/64), 1u);
    k_gemm<1,0,EPI_ACCUM><<<grid,128,0,st>>>(A,B,C,M,N,K,bias,aux);
  }
}
static inline void launch_wconv(hipStream_t st, const float* src, half_t* dst,
                                int K, int N, int Np){
  int tot = K*Np;
  k_wconv<<<(tot+127)/128, 128, 0, st>>>(src, dst, K, N, Np);
}

// workspace layout (bytes)
static constexpr size_t al256(size_t x){ return (x + 255) & ~(size_t)255; }
static constexpr size_t OFF_CONST = 0;
static constexpr size_t OFF_RBF  = al256(OFF_CONST + 256*4);
static constexpr size_t OFF_SB   = al256(OFF_RBF  + (size_t)NE_*6*4);
static constexpr size_t OFF_CTH  = al256(OFF_SB   + (size_t)NE_*42*4);
static constexpr size_t OFF_SBFH = al256(OFF_CTH  + (size_t)NT_*4);
static constexpr size_t OFF_RBFH = al256(OFF_SBFH + (size_t)NT_*8*4);
static constexpr size_t OFF_CAT  = al256(OFF_RBFH + (size_t)NE_*HD*4);
static constexpr size_t OFF_X0   = al256(OFF_CAT  + (size_t)NE_*3*HD*2);
static constexpr size_t OFF_X1   = al256(OFF_X0   + (size_t)NE_*HD*4);
static constexpr size_t OFF_T0   = al256(OFF_X1   + (size_t)NE_*HD*4);
static constexpr size_t OFF_XJ   = al256(OFF_T0   + (size_t)NE_*HD*4);
static constexpr size_t OFF_XK   = al256(OFF_XJ   + (size_t)NE_*HD*4);
static constexpr size_t OFF_TOUT = al256(OFF_XK   + (size_t)NE_*HD*4);
static constexpr size_t OFF_ABIL = al256(OFF_TOUT + (size_t)NE_*HD*4);
static constexpr size_t OFF_WBUF = al256(OFF_ABIL + (size_t)NE_*1024*2);
static constexpr size_t OFF_Y0   = al256(OFF_WBUF + (size_t)1024*HD*2);
static constexpr size_t OFF_Y1   = al256(OFF_Y0   + (size_t)NNODE_*HD*4);
static constexpr size_t OFF_P    = al256(OFF_Y1   + (size_t)NNODE_*HD*4);
// total ~53 MB

extern "C" void kernel_launch(void* const* d_in, const int* in_sizes, int n_in,
                              void* d_out, int out_size, void* d_ws, size_t ws_size,
                              hipStream_t stream){
  (void)out_size; (void)ws_size;
  // ---- input mapping: detect insertion-order vs jax-sorted flattening ----
  const int *p_z, *p_src, *p_dst, *p_ikj, *p_iji;
  const float* p_pos;
  const float* canon[130]; // canonical = reference insertion order of param leaves
  bool sorted = (n_in >= 137) && (in_sizes[1] == NE_);
  if (!sorted){
    p_z   = (const int*)d_in[0];  p_pos = (const float*)d_in[1];
    p_src = (const int*)d_in[3];  p_dst = (const int*)d_in[4];
    p_ikj = (const int*)d_in[5];  p_iji = (const int*)d_in[6];
    for (int i=0;i<130;i++) canon[i] = (const float*)d_in[7+i];
  } else {
    p_dst = (const int*)d_in[1];  p_iji = (const int*)d_in[2];
    p_ikj = (const int*)d_in[3];
    p_pos = (const float*)d_in[134];
    p_src = (const int*)d_in[135]; p_z = (const int*)d_in[136];
    canon[5] = (const float*)d_in[4]; canon[3] = (const float*)d_in[5];
    canon[2] = (const float*)d_in[6]; canon[4] = (const float*)d_in[7];
    canon[1] = (const float*)d_in[8];
    static const int ibmap[21] = {6,14,16,13,15,18,20,17,19,8,10,7,9,12,5,4,3,2,0,1,11};
    for (int ib=0; ib<4; ib++)
      for (int s=0; s<21; s++)
        canon[6 + ib*21 + ibmap[s]] = (const float*)d_in[9 + ib*21 + s];
    static const int obmap[8] = {7,0,1,2,3,4,5,6};
    for (int ob=0; ob<5; ob++)
      for (int s=0; s<8; s++)
        canon[90 + ob*8 + obmap[s]] = (const float*)d_in[93 + ob*8 + s];
    canon[0] = (const float*)d_in[133];
  }
  // canonical layout:
  // 0 rbf_freq; 1 emb.table 2 emb.lin_rbf_w 3 emb.lin_rbf_b 4 emb.lin_w 5 emb.lin_b
  // interaction ib base=6+21*ib: 0 lin_rbf 1 lin_sbf 2 kj_w 3 kj_b 4 ji_w 5 ji_b 6 W
  //   7..10 before{w1,b1,w2,b2} 11 lin_w 12 lin_b 13..20 after0,after1{w1,b1,w2,b2}
  // output ob base=90+8*ob: 0 lin_rbf 1 w0 2 b0 3 w1 4 b1 5 w2 6 b2 7 lin

  char* ws = (char*)d_ws;
  float*  CST  = (float*) (ws + OFF_CONST);
  float*  RBF  = (float*) (ws + OFF_RBF);
  float*  SBm  = (float*) (ws + OFF_SB);
  float*  CTH  = (float*) (ws + OFF_CTH);
  float*  SBFH = (float*) (ws + OFF_SBFH);
  float*  RBFH = (float*) (ws + OFF_RBFH);
  half_t* CATh = (half_t*)(ws + OFF_CAT);
  float*  X0   = (float*) (ws + OFF_X0);
  float*  X1   = (float*) (ws + OFF_X1);
  float*  T0b  = (float*) (ws + OFF_T0);
  float*  XJ   = (float*) (ws + OFF_XJ);
  float*  XK   = (float*) (ws + OFF_XK);
  float*  TOUT = (float*) (ws + OFF_TOUT);
  half_t* ABIL = (half_t*)(ws + OFF_ABIL);
  half_t* WBUF = (half_t*)(ws + OFF_WBUF);
  float*  Y0b  = (float*) (ws + OFF_Y0);
  float*  Y1b  = (float*) (ws + OFF_Y1);
  float*  Pb   = (float*) (ws + OFF_P);

  // ---- constants, geometry ----
  k_init<<<1, 1, 0, stream>>>(CST);
  k_edge<<<(NE_+127)/128, 128, 0, stream>>>(p_pos, p_src, p_dst, canon[0], CST, RBF, SBm);
  k_cth <<<NT_/128, 128, 0, stream>>>(p_pos, p_src, p_dst, p_ikj, p_iji, CTH);

  // ---- embedding: x = silu(cat @ W + b) ----
  k_embcat<<<NE_, 128, 0, stream>>>(RBF, canon[2], canon[3], canon[1], p_z, p_src, p_dst, CATh);
  launch_wconv(stream, canon[4], WBUF, 3*HD, HD, HD);
  launch_gemm(stream, CATh, 1, WBUF, X0, NE_, HD, 3*HD, canon[5], nullptr, EPI_BIAS_SILU);

  k_zero<<<(NNODE_*16+127)/128, 128, 0, stream>>>(Pb, NNODE_*16);

  auto run_output = [&](int ob, const float* xcur){
    const float* const* O = &canon[90 + ob*8];
    k_outt  <<<NE_,    128, 0, stream>>>(RBF, O[0], xcur, TOUT);
    k_gather<<<NNODE_, 128, 0, stream>>>(TOUT, Y0b);
    launch_wconv(stream, O[1], WBUF, HD, HD, HD);
    launch_gemm(stream, Y0b, 0, WBUF, Y1b, NNODE_, HD, HD, O[2], nullptr, EPI_BIAS_SILU);
    launch_wconv(stream, O[3], WBUF, HD, HD, HD);
    launch_gemm(stream, Y1b, 0, WBUF, Y0b, NNODE_, HD, HD, O[4], nullptr, EPI_BIAS_SILU);
    launch_wconv(stream, O[5], WBUF, HD, HD, HD);
    launch_gemm(stream, Y0b, 0, WBUF, Y1b, NNODE_, HD, HD, O[6], nullptr, EPI_BIAS_SILU);
    launch_wconv(stream, O[7], WBUF, HD, OUTD, 16);      // pad 14 -> 16 with zeros
    launch_gemm(stream, Y1b, 0, WBUF, Pb, NNODE_, 16, HD, nullptr, nullptr, EPI_ACCUM);
  };

  run_output(0, X0);

  float* xc = X0;
  float* xn = X1;
  for (int ib=0; ib<4; ib++){
    const float* const* L = &canon[6 + ib*21];
    k_rbfh<<<NE_,    128, 0, stream>>>(RBF, L[0], RBFH);
    k_sbfh<<<NT_/128,128, 0, stream>>>(CTH, p_ikj, SBm, L[1], CST, SBFH);
    // x_kj = silu(x @ kj_w + kj_b) * rbf_h
    launch_wconv(stream, L[2], WBUF, HD, HD, HD);
    launch_gemm(stream, xc, 0, WBUF, XK, NE_, HD, HD, L[3], RBFH, EPI_BIAS_SILU_MUL);
    // per-edge bilinear operand A[e,1024]
    k_abil<<<NE_, 128, 0, stream>>>(XK, SBFH, p_ikj, p_iji, ABIL);
    // x_ji = silu(x @ ji_w + ji_b)
    launch_wconv(stream, L[4], WBUF, HD, HD, HD);
    launch_gemm(stream, xc, 0, WBUF, XJ, NE_, HD, HD, L[5], nullptr, EPI_BIAS_SILU);
    // h = x_ji + A @ Wr   (in-place elementwise add into XJ)
    k_wconv_bil<<<1024, 128, 0, stream>>>(L[6], WBUF);
    launch_gemm(stream, ABIL, 1, WBUF, XJ, NE_, HD, 1024, nullptr, XJ, EPI_ADD_AUX);
    // before-residual: h = h + silu(silu(h@w1+b1)@w2+b2)
    launch_wconv(stream, L[7], WBUF, HD, HD, HD);
    launch_gemm(stream, XJ, 0, WBUF, T0b, NE_, HD, HD, L[8], nullptr, EPI_BIAS_SILU);
    launch_wconv(stream, L[9], WBUF, HD, HD, HD);
    launch_gemm(stream, T0b, 0, WBUF, XJ, NE_, HD, HD, L[10], XJ, EPI_BIAS_SILU_ADD);
    // x_new = silu(h@lin_w+lin_b) + x
    launch_wconv(stream, L[11], WBUF, HD, HD, HD);
    launch_gemm(stream, XJ, 0, WBUF, xn, NE_, HD, HD, L[12], xc, EPI_BIAS_SILU_ADD);
    // two after-residuals (in place on xn)
    for (int rr=0; rr<2; rr++){
      const float* w1 = L[13+rr*4]; const float* b1 = L[14+rr*4];
      const float* w2 = L[15+rr*4]; const float* b2 = L[16+rr*4];
      launch_wconv(stream, w1, WBUF, HD, HD, HD);
      launch_gemm(stream, xn, 0, WBUF, T0b, NE_, HD, HD, b1, nullptr, EPI_BIAS_SILU);
      launch_wconv(stream, w2, WBUF, HD, HD, HD);
      launch_gemm(stream, T0b, 0, WBUF, xn, NE_, HD, HD, b2, xn, EPI_BIAS_SILU_ADD);
    }
    float* tsw = xc; xc = xn; xn = tsw;
    run_output(ib+1, xc);
  }

  k_graphsum<<<4, 128, 0, stream>>>(Pb, (float*)d_out);
}
